// GNN_node_10161892622990
// MI455X (gfx1250) — matrix operand order repaired and checked by reading, not compiled, not run on hardware
//
#include <hip/hip_runtime.h>
#include <hip/hip_bf16.h>
#include <cstdint>

#define NN 50000
#define EE 600000
#define DD 128
#define HH 256
#define LL 5

typedef __attribute__((ext_vector_type(16))) _Float16 v16h;
typedef __attribute__((ext_vector_type(8)))  float    v8f;

union Frag16 { v16h v; uint32_t u[8]; _Float16 h[16]; };
union Half4  { _Float16 h[4]; uint2 u; };

// ---------------------------------------------------------------- init / util
__global__ void init_h_kernel(const int* __restrict__ x_idx,
                              const float* __restrict__ node_emb,
                              float* __restrict__ h) {
    int idx4 = blockIdx.x * 256 + threadIdx.x;        // float4 granularity
    if (idx4 >= NN * DD / 4) return;
    int n = idx4 >> 5, c4 = idx4 & 31;
    ((float4*)h)[idx4] = ((const float4*)node_emb)[x_idx[n] * 32 + c4];
}

__global__ void zero_kernel(float* __restrict__ agg, float* __restrict__ stats) {
    int idx4 = blockIdx.x * 256 + threadIdx.x;
    if (idx4 < NN * DD / 4) ((float4*)agg)[idx4] = make_float4(0.f, 0.f, 0.f, 0.f);
    if (idx4 < 768) stats[idx4] = 0.f;
}

// Pre-transpose per-layer weights into f16 "B-fragment friendly" (col-major) layout.
__global__ void convert_w_kernel(const float* __restrict__ W1l,   // [128,256]
                                 const float* __restrict__ W2l,   // [256,128]
                                 _Float16* __restrict__ W1t,      // [256][128]
                                 _Float16* __restrict__ W2t) {    // [128][256]
    int idx = blockIdx.x * 256 + threadIdx.x;   // 65536 threads
    if (idx < 32768) {
        int n = idx >> 7, k = idx & 127;
        W1t[idx] = (_Float16)W1l[k * HH + n];
    } else {
        int j = idx - 32768;
        int n = j >> 8, k = j & 255;
        W2t[j] = (_Float16)W2l[k * DD + n];
    }
}

// ------------------------------------------- fused edge MLP + gather + scatter
__global__ void __launch_bounds__(256) edge_kernel(
    const int* __restrict__ src, const int* __restrict__ dst,
    const float* __restrict__ eattr, const float* __restrict__ eWl,
    const float* __restrict__ ebl, const float* __restrict__ h,
    float* __restrict__ agg)
{
    __shared__ float s_w[7 * DD];
    __shared__ float s_b[DD];
    for (int t = threadIdx.x; t < 7 * DD; t += 256) s_w[t] = eWl[t];
    if (threadIdx.x < DD) s_b[threadIdx.x] = ebl[threadIdx.x];
    __syncthreads();
    int e = blockIdx.x * 8 + (threadIdx.x >> 5);   // one wave32 per edge
    int lane = threadIdx.x & 31;
    if (e >= EE) return;
    float ea = (lane < 7) ? eattr[(size_t)e * 7 + lane] : 0.f;
    int sn = src[e], dn = dst[e];
    int c0 = lane * 4;                              // 4 channels per lane
    float4 hs = *(const float4*)(h + (size_t)sn * DD + c0);
    float a0 = s_b[c0], a1 = s_b[c0 + 1], a2 = s_b[c0 + 2], a3 = s_b[c0 + 3];
#pragma unroll
    for (int k = 0; k < 7; ++k) {
        float w = __shfl(ea, k, 32);
        const float* wr = s_w + k * DD + c0;
        a0 = fmaf(w, wr[0], a0);
        a1 = fmaf(w, wr[1], a1);
        a2 = fmaf(w, wr[2], a2);
        a3 = fmaf(w, wr[3], a3);
    }
    float* ap = agg + (size_t)dn * DD + c0;
    atomicAdd(ap + 0, fmaxf(hs.x + a0, 0.f));
    atomicAdd(ap + 1, fmaxf(hs.y + a1, 0.f));
    atomicAdd(ap + 2, fmaxf(hs.z + a2, 0.f));
    atomicAdd(ap + 3, fmaxf(hs.w + a3, 0.f));
}

// z = (1+eps)*h + agg, converted to f16 for WMMA A-operand (float4 vectorized)
__global__ void combine_kernel(const float* __restrict__ h, const float* __restrict__ agg,
                               const float* __restrict__ epsp, int l,
                               _Float16* __restrict__ zf16) {
    int idx4 = blockIdx.x * 256 + threadIdx.x;
    if (idx4 >= NN * DD / 4) return;
    float e = 1.f + epsp[l];
    float4 hv = ((const float4*)h)[idx4];
    float4 av = ((const float4*)agg)[idx4];
    Half4 o;
    o.h[0] = (_Float16)fmaf(e, hv.x, av.x);
    o.h[1] = (_Float16)fmaf(e, hv.y, av.y);
    o.h[2] = (_Float16)fmaf(e, hv.z, av.z);
    o.h[3] = (_Float16)fmaf(e, hv.w, av.w);
    *(uint2*)(zf16 + (size_t)idx4 * 4) = o.u;
}

// ---------------- GEMM1: [N,128]x[128,256], 16x64 tile per wave (A reused 4x)
__global__ void __launch_bounds__(256) gemm1_kernel(
    const _Float16* __restrict__ zf16, const _Float16* __restrict__ W1t,
    const float* __restrict__ bias1, float* __restrict__ y,
    float* __restrict__ sum1, float* __restrict__ sq1)
{
    int wid = (blockIdx.x * 256 + threadIdx.x) >> 5;
    int lane = threadIdx.x & 31;
    if (wid >= (NN / 16) * 4) return;               // 3125 row tiles x 4 col groups
    int rowTile = wid >> 2;
    int cg = wid & 3;
    int l15 = lane & 15;
    int kh = lane >> 4;
    int r = rowTile * 16 + l15;
    int n0 = cg * 64 + l15;
    const _Float16* zrow = zf16 + (size_t)r * DD;
    v8f acc[4] = {};
#pragma unroll
    for (int kk = 0; kk < 4; ++kk) {                // K = 128 = 4 x 32
        int kb = kk * 32;
        Frag16 a;
#pragma unroll
        for (int v = 0; v < 8; ++v) {
            int k0 = kb + 16 * kh + 2 * v;
            a.u[v] = *(const uint32_t*)(zrow + k0);
        }
#pragma unroll
        for (int t = 0; t < 4; ++t) {
            Frag16 b;
            const uint4* bp =
                (const uint4*)(W1t + (size_t)(n0 + 16 * t) * DD + kb + 16 * kh);
            uint4 q0 = bp[0], q1 = bp[1];
            b.u[0] = q0.x; b.u[1] = q0.y; b.u[2] = q0.z; b.u[3] = q0.w;
            b.u[4] = q1.x; b.u[5] = q1.y; b.u[6] = q1.z; b.u[7] = q1.w;
            acc[t] = __builtin_amdgcn_wmma_f32_16x16x32_f16(false, a.v, false, b.v,
                                                            (short)0, acc[t],
                                                            false, false);
        }
    }
#pragma unroll
    for (int t = 0; t < 4; ++t) {
        int n = n0 + 16 * t;
        float bias = bias1[n];
        float s = 0.f, s2 = 0.f;
#pragma unroll
        for (int i = 0; i < 8; ++i) {
            int m = rowTile * 16 + 8 * kh + i;
            float val = acc[t][i] + bias;
            y[(size_t)m * HH + n] = val;
            s += val; s2 += val * val;
        }
        atomicAdd(&sum1[n], s);   // fused BN1 batch statistics
        atomicAdd(&sq1[n], s2);
    }
}

// fold mean/var/gamma/beta into per-channel scale/shift
__global__ void bn_params_kernel(int C, float invN,
                                 const float* __restrict__ sum, const float* __restrict__ sq,
                                 const float* __restrict__ g, const float* __restrict__ b,
                                 float* __restrict__ scale, float* __restrict__ shift) {
    int c = threadIdx.x;
    if (c >= C) return;
    float m = sum[c] * invN;
    float var = sq[c] * invN - m * m;
    float sc = g[c] * rsqrtf(var + 1e-5f);
    scale[c] = sc;
    shift[c] = fmaf(-m, sc, b[c]);
}

// -- GEMM2: relu(bn(y)) [N,256] x [256,128], 16x64 tile/wave, BN fused in A-load
__global__ void __launch_bounds__(256) gemm2_kernel(
    const float* __restrict__ y, const _Float16* __restrict__ W2t,
    const float* __restrict__ scale1, const float* __restrict__ shift1,
    const float* __restrict__ bias2, float* __restrict__ z2,
    float* __restrict__ sum2, float* __restrict__ sq2)
{
    __shared__ float s_sc[HH], s_sh[HH];
    s_sc[threadIdx.x] = scale1[threadIdx.x];
    s_sh[threadIdx.x] = shift1[threadIdx.x];
    __syncthreads();
    int wid = (blockIdx.x * 256 + threadIdx.x) >> 5;
    int lane = threadIdx.x & 31;
    if (wid >= (NN / 16) * 2) return;               // 3125 row tiles x 2 col groups
    int rowTile = wid >> 1;
    int cg = wid & 1;
    int l15 = lane & 15;
    int kh = lane >> 4;
    int r = rowTile * 16 + l15;
    int n0 = cg * 64 + l15;
    const float* yrow = y + (size_t)r * HH;
    v8f acc[4] = {};
#pragma unroll
    for (int kk = 0; kk < 8; ++kk) {                // K = 256 = 8 x 32
        int kb = kk * 32;
        Frag16 a;
#pragma unroll
        for (int v = 0; v < 8; ++v) {               // BN+ReLU+cvt fused A-prep
            int k0 = kb + 16 * kh + 2 * v;
            float2 f = *(const float2*)(yrow + k0);
            float x0 = fmaxf(fmaf(f.x, s_sc[k0], s_sh[k0]), 0.f);
            float x1 = fmaxf(fmaf(f.y, s_sc[k0 + 1], s_sh[k0 + 1]), 0.f);
            a.h[2 * v]     = (_Float16)x0;
            a.h[2 * v + 1] = (_Float16)x1;
        }
#pragma unroll
        for (int t = 0; t < 4; ++t) {
            Frag16 b;
            const uint4* bp =
                (const uint4*)(W2t + (size_t)(n0 + 16 * t) * HH + kb + 16 * kh);
            uint4 q0 = bp[0], q1 = bp[1];
            b.u[0] = q0.x; b.u[1] = q0.y; b.u[2] = q0.z; b.u[3] = q0.w;
            b.u[4] = q1.x; b.u[5] = q1.y; b.u[6] = q1.z; b.u[7] = q1.w;
            acc[t] = __builtin_amdgcn_wmma_f32_16x16x32_f16(false, a.v, false, b.v,
                                                            (short)0, acc[t],
                                                            false, false);
        }
    }
#pragma unroll
    for (int t = 0; t < 4; ++t) {
        int n = n0 + 16 * t;
        float bias = bias2[n];
        float s = 0.f, s2 = 0.f;
#pragma unroll
        for (int i = 0; i < 8; ++i) {
            int m = rowTile * 16 + 8 * kh + i;
            float val = acc[t][i] + bias;
            z2[(size_t)m * DD + n] = val;
            s += val; s2 += val * val;
        }
        atomicAdd(&sum2[n], s);   // fused outer-BN batch statistics
        atomicAdd(&sq2[n], s2);
    }
}

// outer BN apply (+relu except last layer), float4 vectorized
__global__ void apply_out_kernel(const float* __restrict__ z2,
                                 const float* __restrict__ scale,
                                 const float* __restrict__ shift,
                                 float* __restrict__ out, int doRelu) {
    int idx4 = blockIdx.x * 256 + threadIdx.x;
    if (idx4 >= NN * DD / 4) return;
    int c0 = (idx4 * 4) & 127;
    float4 z = ((const float4*)z2)[idx4];
    float4 o;
    o.x = fmaf(z.x, scale[c0 + 0], shift[c0 + 0]);
    o.y = fmaf(z.y, scale[c0 + 1], shift[c0 + 1]);
    o.z = fmaf(z.z, scale[c0 + 2], shift[c0 + 2]);
    o.w = fmaf(z.w, scale[c0 + 3], shift[c0 + 3]);
    if (doRelu) {
        o.x = fmaxf(o.x, 0.f); o.y = fmaxf(o.y, 0.f);
        o.z = fmaxf(o.z, 0.f); o.w = fmaxf(o.w, 0.f);
    }
    ((float4*)out)[idx4] = o;
}

// ---------------------------------------------------------------------- host
extern "C" void kernel_launch(void* const* d_in, const int* in_sizes, int n_in,
                              void* d_out, int out_size, void* d_ws, size_t ws_size,
                              hipStream_t stream) {
    (void)in_sizes; (void)n_in; (void)out_size; (void)ws_size;
    const int*   x_idx    = (const int*)d_in[0];
    const int*   eidx     = (const int*)d_in[1];
    const float* eattr    = (const float*)d_in[2];
    const float* node_emb = (const float*)d_in[3];
    const float* edge_W   = (const float*)d_in[4];
    const float* edge_b   = (const float*)d_in[5];
    const float* W1       = (const float*)d_in[6];
    const float* b1       = (const float*)d_in[7];
    const float* bn1g     = (const float*)d_in[8];
    const float* bn1b     = (const float*)d_in[9];
    const float* W2       = (const float*)d_in[10];
    const float* b2       = (const float*)d_in[11];
    const float* epsp     = (const float*)d_in[12];
    const float* obng     = (const float*)d_in[13];
    const float* obnb     = (const float*)d_in[14];
    const int* srcp = eidx;
    const int* dstp = eidx + EE;

    char* ws = (char*)d_ws;
    float*    h    = (float*)(ws);                      // [N,128] f32
    float*    agg  = (float*)(ws + 25600000);           // [N,128] f32
    _Float16* zf16 = (_Float16*)(ws + 51200000);        // [N,128] f16
    float*    y    = (float*)(ws + 64000000);           // [N,256] f32
    float*    z2   = (float*)(ws + 115200000);          // [N,128] f32
    _Float16* W1t  = (_Float16*)(ws + 140800000);       // [256][128] f16
    _Float16* W2t  = (_Float16*)(ws + 140865536);       // [128][256] f16
    float*    stats = (float*)(ws + 140931072);
    float* sum1 = stats;          float* sq1  = stats + 256;
    float* sum2 = stats + 512;    float* sq2  = stats + 640;
    float* scale1 = stats + 768;  float* shift1 = stats + 1024;
    float* scale2 = stats + 1280; float* shift2 = stats + 1408;

    init_h_kernel<<<6250, 256, 0, stream>>>(x_idx, node_emb, h);

    for (int l = 0; l < LL; ++l) {
        convert_w_kernel<<<256, 256, 0, stream>>>(W1 + l * DD * HH, W2 + l * HH * DD,
                                                  W1t, W2t);
        zero_kernel<<<6250, 256, 0, stream>>>(agg, stats);
        edge_kernel<<<EE / 8, 256, 0, stream>>>(srcp, dstp, eattr,
                                                edge_W + l * 7 * DD, edge_b + l * DD,
                                                h, agg);
        combine_kernel<<<6250, 256, 0, stream>>>(h, agg, epsp, l, zf16);
        gemm1_kernel<<<1563, 256, 0, stream>>>(zf16, W1t, b1 + l * HH, y, sum1, sq1);
        bn_params_kernel<<<1, 256, 0, stream>>>(HH, 1.f / NN, sum1, sq1,
                                                bn1g + l * HH, bn1b + l * HH,
                                                scale1, shift1);
        gemm2_kernel<<<782, 256, 0, stream>>>(y, W2t, scale1, shift1, b2 + l * DD,
                                              z2, sum2, sq2);
        bn_params_kernel<<<1, 256, 0, stream>>>(DD, 1.f / NN, sum2, sq2,
                                                obng + l * DD, obnb + l * DD,
                                                scale2, shift2);
        float* outp = (l == LL - 1) ? (float*)d_out : h;
        apply_out_kernel<<<6250, 256, 0, stream>>>(z2, scale2, shift2, outp,
                                                   (l < LL - 1) ? 1 : 0);
    }
}
